// TalkingMultiheadSelfAttention_18459769438416
// MI455X (gfx1250) — matrix operand order repaired and checked by reading, not compiled, not run on hardware
//
#include <hip/hip_runtime.h>
#include <hip/hip_bf16.h>

// ---------------------------------------------------------------------------
// Talking-heads multihead self-attention for gfx1250 (CDNA5, wave32, WMMA).
//   B=64, N=256, C=512, H=8, D=64
// All GEMMs use v_wmma_f32_16x16x32_f16 with 16x64 per-wave tiles (A-fragment
// reused across 4 B-fragments -> ~13 FLOP/B fragment intensity).
// Fragment layouts (ISA 7.12.2, wave32):
//   A 16x32 f16: lane m=L%16; h0..15 = K(L<16?0..15:16..31) (contiguous; code's order, not ISA table)
//   B 32x16 f16: lane n=L%16; h0..15 = K(L<16?0..15:16..31)  (contiguous 32B; code's order, not ISA's)
//   C/D f32:     lane n=L%16; vgpr j -> row m = j + 8*(L>=16)
// ---------------------------------------------------------------------------

typedef __attribute__((ext_vector_type(16))) _Float16 v16h;
typedef __attribute__((ext_vector_type(8)))  float    v8f;

#define Bv   64
#define Nv   256
#define Cv   512
#define Hv   8
#define Dv   64

union FragU { uint4 u[2]; v16h h; };

__device__ __forceinline__ v16h frag_split(const _Float16* lo, const _Float16* hi) {
    FragU f;
    f.u[0] = *reinterpret_cast<const uint4*>(lo);
    f.u[1] = *reinterpret_cast<const uint4*>(hi);
    return f.h;
}
__device__ __forceinline__ v16h frag_contig(const _Float16* p) {
    return frag_split(p, p + 8);
}
// A fragment from f16 row-major: contiguous 16 K per lane half (off = 0/16)
__device__ __forceinline__ v16h frag_A(const _Float16* row, int hif) {
    const _Float16* p = row + (hif ? 16 : 0);
    return frag_split(p, p + 8);
}

// ---------------------------------------------------------------------------
// Kernel 0: f16 conversion of inputs + transposed f16 weights.
//   Xh [16384][512], WqkvT [1536][512], WoutT [512][512]
// ---------------------------------------------------------------------------
__global__ void __launch_bounds__(256) tmha_prep_kernel(
        const float* __restrict__ X, const float* __restrict__ Wqkv,
        const float* __restrict__ Wout,
        _Float16* __restrict__ Xh, _Float16* __restrict__ WqkvT,
        _Float16* __restrict__ WoutT) {
    int idx = blockIdx.x * 256 + threadIdx.x;
    if (idx < Bv * Nv * Cv)                       // 8,388,608
        Xh[idx] = (_Float16)X[idx];
    if (idx < 3 * Cv * Cv) {                      // 786,432
        int n = idx >> 9, k = idx & 511;
        WqkvT[idx] = (_Float16)Wqkv[k * (3 * Cv) + n];
    }
    if (idx < Cv * Cv) {                          // 262,144
        int n = idx >> 9, k = idx & 511;
        WoutT[idx] = (_Float16)Wout[k * Cv + n];
    }
}

// ---------------------------------------------------------------------------
// Kernel 1: QKV projection. One wave per 16x64 tile of (16384 x 1536).
// A 64-wide column group lies inside a single (s, head) pair, so the
// layout-scatter to Q (B,H,N,D), K (B,H,N,D), Vt (B,H,D,N) is branch-uniform.
// ---------------------------------------------------------------------------
__global__ void __launch_bounds__(256) tmha_qkv_kernel(
        const _Float16* __restrict__ Xh, const _Float16* __restrict__ WqkvT,
        const float* __restrict__ bqkv,
        _Float16* __restrict__ Q, _Float16* __restrict__ Kd,
        _Float16* __restrict__ Vt) {
    const int lane = threadIdx.x & 31;
    const int wv   = threadIdx.x >> 5;
    const int tile = blockIdx.x * 8 + wv;          // 24576 tiles (1024 x 24)
    const int mt = tile / 24, cg = tile % 24;      // 16-row block, 64-col group
    const int m16 = lane & 15, hif = lane >> 4;

    const _Float16* arow = Xh + (size_t)(mt * 16 + m16) * Cv;
    const _Float16* brow0 = WqkvT + (size_t)(cg * 64 + m16) * Cv + (hif ? 16 : 0);

    v8f acc[4] = {};
#pragma unroll
    for (int kk = 0; kk < 16; ++kk) {              // K = 512 in steps of 32
        v16h a = frag_A(arow + kk * 32, hif);
#pragma unroll
        for (int nt = 0; nt < 4; ++nt) {
            v16h b = frag_contig(brow0 + (size_t)nt * 16 * Cv + kk * 32);
            acc[nt] = __builtin_amdgcn_wmma_f32_16x16x32_f16(false, a, false, b,
                                                             (short)0, acc[nt], false, false);
        }
    }
    const int s  = cg >> 3;                        // 0=q 1=k 2=v (uniform per wave)
    const int hh = cg & 7;
#pragma unroll
    for (int nt = 0; nt < 4; ++nt) {
        const int d    = nt * 16 + m16;
        const float bias = bqkv[cg * 64 + d];
#pragma unroll
        for (int j = 0; j < 8; ++j) {
            const int row  = mt * 16 + j + hif * 8;
            const int b_   = row >> 8, nseq = row & 255;
            const _Float16 v = (_Float16)(acc[nt][j] + bias);
            if (s == 0)      Q [((b_ * Hv + hh) * Nv + nseq) * Dv + d] = v;
            else if (s == 1) Kd[((b_ * Hv + hh) * Nv + nseq) * Dv + d] = v;
            else             Vt[((b_ * Hv + hh) * Dv + d) * Nv + nseq] = v;
        }
    }
}

// ---------------------------------------------------------------------------
// Kernel 2: fused attention. Block = 8 waves (one per head) per (b, q-tile).
// 128KB dynamic LDS slab for the two cross-head talking-heads mixes.
// ---------------------------------------------------------------------------
__global__ void __launch_bounds__(256) tmha_attn_kernel(
        const _Float16* __restrict__ Q, const _Float16* __restrict__ Kd,
        const _Float16* __restrict__ Vt,
        const float* __restrict__ inter, const unsigned char* __restrict__ mask,
        const float* __restrict__ Wt1, const float* __restrict__ bt1,
        const float* __restrict__ Wt2, const float* __restrict__ bt2,
        _Float16* __restrict__ AO) {
    extern __shared__ char smem[];
    float*    S  = reinterpret_cast<float*>(smem);      // [8][16][256] f32
    _Float16* Ah = reinterpret_cast<_Float16*>(smem);   // reused [8][16][256] f16

    const int lane = threadIdx.x & 31;
    const int g    = threadIdx.x >> 5;                  // head / wave id
    const int m16  = lane & 15, hif = lane >> 4;
    const int b    = blockIdx.x >> 4;
    const int qt   = blockIdx.x & 15;

    // prefetch this block's contiguous interaction slab (16*256*8 f32 = 128KB)
    // so the HBM-bound read overlaps phase-1 WMMAs (global_prefetch_b8).
    {
        const char* ib = reinterpret_cast<const char*>(
            inter + (size_t)(b * Nv + qt * 16) * Nv * Hv);
#pragma unroll
        for (int i = 0; i < 4; ++i)
            __builtin_prefetch(ib + ((size_t)threadIdx.x + i * 256) * 128, 0, 3);
    }

    // ---- phase 1: S_g = (Q_g K_g^T) / 8 ------------------------------------
    const _Float16* qrow = Q + ((size_t)(b * Hv + g) * Nv + qt * 16 + m16) * Dv;
    const _Float16* kbase = Kd + (size_t)(b * Hv + g) * Nv * Dv;
    v16h aq0 = frag_A(qrow, hif);                       // kt-invariant A frags
    v16h aq1 = frag_A(qrow + 32, hif);
#pragma unroll
    for (int kt = 0; kt < 16; ++kt) {
        const _Float16* kb = kbase + (size_t)(kt * 16 + m16) * Dv + (hif ? 16 : 0);
        v8f acc = {};
        acc = __builtin_amdgcn_wmma_f32_16x16x32_f16(false, aq0, false,
                  frag_contig(kb), (short)0, acc, false, false);
        acc = __builtin_amdgcn_wmma_f32_16x16x32_f16(false, aq1, false,
                  frag_contig(kb + 32), (short)0, acc, false, false);
#pragma unroll
        for (int j = 0; j < 8; ++j)
            S[(g * 16 + j + hif * 8) * Nv + kt * 16 + m16] = acc[j] * 0.125f;
    }
    __syncthreads();

    // ---- phase 2: mix1 + interaction + mask + softmax + mix2 ---------------
    float w1[8], w2[8];
#pragma unroll
    for (int hh = 0; hh < 8; ++hh) { w1[hh] = Wt1[hh * 8 + g]; w2[hh] = Wt2[hh * 8 + g]; }
    const float b1 = bt1[g], b2 = bt2[g];

    float val[16][8];                                   // c = cc*32 + lane
#pragma unroll
    for (int r = 0; r < 16; ++r) {
        const size_t rowb = ((size_t)b * Nv + qt * 16 + r) * Nv;
#pragma unroll
        for (int cc = 0; cc < 8; ++cc) {
            const int c = cc * 32 + lane;
            float s = b1;
#pragma unroll
            for (int hh = 0; hh < 8; ++hh) s += S[(hh * 16 + r) * Nv + c] * w1[hh];
            s += inter[(rowb + c) * Hv + g];
            s += (1.0f - (float)mask[rowb + c]) * -1e9f;
            val[r][cc] = s;
        }
    }
#pragma unroll
    for (int r = 0; r < 16; ++r) {                      // softmax over keys
        float mx = val[r][0];
#pragma unroll
        for (int cc = 1; cc < 8; ++cc) mx = fmaxf(mx, val[r][cc]);
#pragma unroll
        for (int off = 16; off >= 1; off >>= 1) mx = fmaxf(mx, __shfl_xor(mx, off, 32));
        float sum = 0.f;
#pragma unroll
        for (int cc = 0; cc < 8; ++cc) { val[r][cc] = __expf(val[r][cc] - mx); sum += val[r][cc]; }
#pragma unroll
        for (int off = 16; off >= 1; off >>= 1) sum += __shfl_xor(sum, off, 32);
        const float inv = 1.0f / sum;
#pragma unroll
        for (int cc = 0; cc < 8; ++cc) val[r][cc] *= inv;
    }
    __syncthreads();                                    // S reads done
#pragma unroll
    for (int r = 0; r < 16; ++r)
#pragma unroll
        for (int cc = 0; cc < 8; ++cc) S[(g * 16 + r) * Nv + cc * 32 + lane] = val[r][cc];
    __syncthreads();
#pragma unroll
    for (int r = 0; r < 16; ++r)
#pragma unroll
        for (int cc = 0; cc < 8; ++cc) {
            const int c = cc * 32 + lane;
            float s = b2;
#pragma unroll
            for (int hh = 0; hh < 8; ++hh) s += S[(hh * 16 + r) * Nv + c] * w2[hh];
            val[r][cc] = s;
        }
    __syncthreads();                                    // reuse slab as f16
#pragma unroll
    for (int r = 0; r < 16; ++r)
#pragma unroll
        for (int cc = 0; cc < 8; ++cc)
            Ah[(g * 16 + r) * Nv + cc * 32 + lane] = (_Float16)val[r][cc];
    __syncthreads();

    // ---- phase 3: out_g = attn2_g (16x256) @ V_g (256x64) ------------------
    v8f oacc[4] = {};
    const _Float16* vbase = Vt + (size_t)(b * Hv + g) * Dv * Nv;
#pragma unroll
    for (int kt = 0; kt < 8; ++kt) {                    // 256 keys = 8x32
        v16h a = frag_A(Ah + (g * 16 + m16) * Nv + kt * 32, hif);
#pragma unroll
        for (int nt = 0; nt < 4; ++nt) {
            v16h bf = frag_contig(vbase + (size_t)(nt * 16 + m16) * Nv
                                        + kt * 32 + (hif ? 16 : 0));
            oacc[nt] = __builtin_amdgcn_wmma_f32_16x16x32_f16(false, a, false, bf,
                                                              (short)0, oacc[nt], false, false);
        }
    }
#pragma unroll
    for (int nt = 0; nt < 4; ++nt)
#pragma unroll
        for (int j = 0; j < 8; ++j)
            AO[((size_t)(b * Nv + qt * 16 + j + hif * 8)) * Cv
               + g * Dv + nt * 16 + m16] = (_Float16)oacc[nt][j];
}

// ---------------------------------------------------------------------------
// Kernel 3: output projection. One wave per 16x64 tile of (16384 x 512).
// ---------------------------------------------------------------------------
__global__ void __launch_bounds__(256) tmha_proj_kernel(
        const _Float16* __restrict__ AO, const _Float16* __restrict__ WoutT,
        const float* __restrict__ bout, float* __restrict__ out) {
    const int lane = threadIdx.x & 31;
    const int wv   = threadIdx.x >> 5;
    const int tile = blockIdx.x * 8 + wv;               // 8192 tiles (1024 x 8)
    const int mt = tile >> 3, cg = tile & 7;
    const int m16 = lane & 15, hif = lane >> 4;

    const _Float16* arow  = AO + (size_t)(mt * 16 + m16) * Cv;
    const _Float16* brow0 = WoutT + (size_t)(cg * 64 + m16) * Cv + (hif ? 16 : 0);

    v8f acc[4] = {};
#pragma unroll
    for (int kk = 0; kk < 16; ++kk) {
        v16h a = frag_A(arow + kk * 32, hif);
#pragma unroll
        for (int nt = 0; nt < 4; ++nt) {
            v16h b = frag_contig(brow0 + (size_t)nt * 16 * Cv + kk * 32);
            acc[nt] = __builtin_amdgcn_wmma_f32_16x16x32_f16(false, a, false, b,
                                                             (short)0, acc[nt], false, false);
        }
    }
#pragma unroll
    for (int nt = 0; nt < 4; ++nt) {
        const int col = cg * 64 + nt * 16 + m16;
        const float bias = bout[col];
#pragma unroll
        for (int j = 0; j < 8; ++j)
            out[(size_t)(mt * 16 + j + hif * 8) * Cv + col] = acc[nt][j] + bias;
    }
}

// ---------------------------------------------------------------------------
extern "C" void kernel_launch(void* const* d_in, const int* in_sizes, int n_in,
                              void* d_out, int out_size, void* d_ws, size_t ws_size,
                              hipStream_t stream) {
    const float* X      = (const float*)d_in[0];
    const unsigned char* mask = (const unsigned char*)d_in[1];
    const float* inter  = (const float*)d_in[2];
    const float* Wqkv   = (const float*)d_in[3];
    const float* bqkv   = (const float*)d_in[4];
    const float* Wt1    = (const float*)d_in[5];
    const float* bt1    = (const float*)d_in[6];
    const float* Wt2    = (const float*)d_in[7];
    const float* bt2    = (const float*)d_in[8];
    const float* Wout   = (const float*)d_in[9];
    const float* bout   = (const float*)d_in[10];
    float* out = (float*)d_out;

    // workspace carve-up (f16 elements): 5x 8.4M + 0.79M + 0.26M  (~86 MB)
    _Float16* ws    = (_Float16*)d_ws;
    const size_t nQKV = (size_t)Bv * Hv * Nv * Dv;      // 8,388,608 each
    _Float16* Q     = ws;
    _Float16* Kd    = Q  + nQKV;
    _Float16* Vt    = Kd + nQKV;
    _Float16* AO    = Vt + nQKV;
    _Float16* Xh    = AO + nQKV;
    _Float16* WqkvT = Xh + nQKV;
    _Float16* WoutT = WqkvT + (size_t)3 * Cv * Cv;

    tmha_prep_kernel<<<32768, 256, 0, stream>>>(X, Wqkv, Wout, Xh, WqkvT, WoutT);
    tmha_qkv_kernel<<<3072, 256, 0, stream>>>(Xh, WqkvT, bqkv, Q, Kd, Vt);
    tmha_attn_kernel<<<Bv * (Nv / 16), 256, Hv * 16 * Nv * sizeof(float), stream>>>(
        Q, Kd, Vt, inter, mask, Wt1, bt1, Wt2, bt2, AO);
    tmha_proj_kernel<<<1024, 256, 0, stream>>>(AO, WoutT, bout, out);
}
